// SNN_Model_29162827940630
// MI455X (gfx1250) — compile-verified, matrix operand order checked by reading, not run on hardware
//
#include <hip/hip_runtime.h>
#include <hip/hip_bf16.h>
#include <cmath>

typedef __attribute__((ext_vector_type(16))) __bf16 v16bf;
typedef __attribute__((ext_vector_type(8)))  float  v8f;

namespace {
constexpr int IN   = 784;
constexpr int KP   = 800;   // layer-1 K padded to multiple of 32
constexpr int H    = 512;
constexpr int OUT  = 10;
constexpr int NP   = 16;    // OUT padded to 16 for WMMA
constexpr int BT   = 8192;  // batch
constexpr float THRESH = 0.35f;
constexpr float WDEC   = 0.95f;
}

// ---- fragment-ready swizzled storage --------------------------------------
// For a 16(outer) x 32(K) WMMA tile, lane l needs 16 contiguous bf16:
//   lane = (outer&15) | ((k>>3 & 1) << 4) ; elem = (k&7) | ((k>>4 & 1) << 3)
// Identical mapping for A (outer=M) and B (outer=N) operands.
// Storage: tile-major, tile = (outer/16)*Ktiles + (k/32); 512 elems per tile.
__device__ __forceinline__ size_t frag_off(int outer, int k, int ktiles) {
  size_t tile = (size_t)(outer >> 4) * ktiles + (k >> 5);
  int l = (outer & 15) | (((k >> 3) & 1) << 4);
  int e = (k & 7) | (((k >> 4) & 1) << 3);
  return tile * 512 + (size_t)l * 16 + e;
}

// One aligned 32-byte load -> 2x global_load_b128, no assembly VALU.
__device__ __forceinline__ v16bf load_frag(const __bf16* __restrict__ base,
                                           int outer0, int k0, int ktiles,
                                           int lane) {
  size_t tile = (size_t)(outer0 >> 4) * ktiles + (k0 >> 5);
  return *(const v16bf*)(base + tile * 512 + (size_t)lane * 16);
}

// ---- per-timestep prep ----------------------------------------------------
// xbf = xf (bf16, K padded, A-swizzled for gemm1)
// pre1T[k-dim=batch, outer=input-idx] = xf*beta1 (A-swizzled for hebb1)
__global__ void __launch_bounds__(256)
prep_x(const float* __restrict__ x, const float* __restrict__ beta1,
       __bf16* __restrict__ xbf, __bf16* __restrict__ pre1T, float decay) {
  int idx = blockIdx.x * 256 + threadIdx.x;       // over BT*KP
  int b = idx / KP, k = idx % KP;
  float v = (k < IN) ? x[(size_t)b * IN + k] * decay : 0.0f;
  xbf[frag_off(b, k, KP / 32)] = (__bf16)v;
  if (k < IN) pre1T[frag_off(k, b, BT / 32)] = (__bf16)(v * beta1[k]);
}

// W1eff[k,n] = W1[n,k] + alpha1*hb1[k,n]  (B-swizzled, rows >=IN zeroed)
__global__ void __launch_bounds__(256)
weff1(const float* __restrict__ W1, const float* __restrict__ hb1f,
      const float* __restrict__ alpha1, __bf16* __restrict__ w1e) {
  int idx = blockIdx.x * 256 + threadIdx.x;       // over KP*H
  int k = idx / H, n = idx % H;
  float v = 0.0f;
  if (k < IN) v = W1[(size_t)n * IN + k] + alpha1[0] * hb1f[(size_t)k * H + n];
  w1e[frag_off(n, k, KP / 32)] = (__bf16)v;
}

// W2eff[k,n] = W2[n,k] + alpha2*hb2[k,n]  (B-swizzled, cols >=OUT zeroed)
__global__ void __launch_bounds__(256)
weff2(const float* __restrict__ W2, const float* __restrict__ hb2f,
      const float* __restrict__ alpha2, __bf16* __restrict__ w2e) {
  int idx = blockIdx.x * 256 + threadIdx.x;       // over H*NP
  int k = idx / NP, n = idx % NP;
  float v = 0.0f;
  if (n < OUT) v = W2[(size_t)n * H + k] + alpha2[0] * hb2f[(size_t)k * OUT + n];
  w2e[frag_off(n, k, H / 32)] = (__bf16)v;
}

// ---- layer 1: h1 = xf @ W1eff + b1, fused LIF + traces --------------------
// Each wave: 64x16 output (4 M-tiles reusing one B fragment per K-step).
__global__ void __launch_bounds__(256)
gemm1_lif(const __bf16* __restrict__ xbf, const __bf16* __restrict__ w1e,
          const float* __restrict__ b1, const float* __restrict__ eta1,
          float* __restrict__ v1, float* __restrict__ s1,
          __bf16* __restrict__ sp1, __bf16* __restrict__ post1) {
  const int lane = threadIdx.x & 31;
  const int wave = blockIdx.x * 8 + (threadIdx.x >> 5);
  const int NT = H / 16;                          // 32
  const int m0 = (wave / NT) * 64, n0 = (wave % NT) * 16;
  v8f c0 = {}, c1 = {}, c2 = {}, c3 = {};
  for (int k0 = 0; k0 < KP; k0 += 32) {
    v16bf bf = load_frag(w1e, n0, k0, KP / 32, lane);
    v16bf a0 = load_frag(xbf, m0,      k0, KP / 32, lane);
    v16bf a1 = load_frag(xbf, m0 + 16, k0, KP / 32, lane);
    v16bf a2 = load_frag(xbf, m0 + 32, k0, KP / 32, lane);
    v16bf a3 = load_frag(xbf, m0 + 48, k0, KP / 32, lane);
    c0 = __builtin_amdgcn_wmma_f32_16x16x32_bf16(false, a0, false, bf, (short)0, c0, false, false);
    c1 = __builtin_amdgcn_wmma_f32_16x16x32_bf16(false, a1, false, bf, (short)0, c1, false, false);
    c2 = __builtin_amdgcn_wmma_f32_16x16x32_bf16(false, a2, false, bf, (short)0, c2, false, false);
    c3 = __builtin_amdgcn_wmma_f32_16x16x32_bf16(false, a3, false, bf, (short)0, c3, false, false);
  }
  const int n    = n0 + (lane & 15);
  const int moff = (lane & 16) ? 8 : 0;
  const float bias = b1[n];
  const float eta  = eta1[n];
#define EPILOG1(C, MB)                                                   \
  _Pragma("unroll") for (int r = 0; r < 8; ++r) {                        \
    int m = (MB) + r + moff;                                             \
    float h = (C)[r] + bias;                                             \
    size_t off = (size_t)m * H + n;                                      \
    float v = v1[off];                                                   \
    v = 0.5f * (v + h);                    /* tau=2 charge */            \
    float sp = (v >= 1.0f) ? 1.0f : 0.0f;  /* spike */                   \
    v1[off] = (1.0f - sp) * v;             /* hard reset */              \
    s1[off] += sp;                                                       \
    sp1[frag_off(m, n, H / 32)] = (__bf16)sp;       /* A of gemm2 */     \
    post1[frag_off(n, m, BT / 32)] =                /* B of hebb1 */     \
        (__bf16)tanhf(h * (1.0f / THRESH) - eta);                        \
  }
  EPILOG1(c0, m0) EPILOG1(c1, m0 + 16) EPILOG1(c2, m0 + 32) EPILOG1(c3, m0 + 48)
#undef EPILOG1
}

// ---- layer 2: h2 = sp1 @ W2eff + b2, fused LIF (N padded to 16) -----------
__global__ void __launch_bounds__(256)
gemm2_lif(const __bf16* __restrict__ sp1, const __bf16* __restrict__ w2e,
          const float* __restrict__ b2, const float* __restrict__ eta2,
          float* __restrict__ v2, float* __restrict__ s2,
          __bf16* __restrict__ post2) {
  const int lane = threadIdx.x & 31;
  const int wave = blockIdx.x * 8 + (threadIdx.x >> 5);
  const int m0 = wave * 64;
  v8f c0 = {}, c1 = {}, c2 = {}, c3 = {};
  for (int k0 = 0; k0 < H; k0 += 32) {
    v16bf bf = load_frag(w2e, 0, k0, H / 32, lane);
    v16bf a0 = load_frag(sp1, m0,      k0, H / 32, lane);
    v16bf a1 = load_frag(sp1, m0 + 16, k0, H / 32, lane);
    v16bf a2 = load_frag(sp1, m0 + 32, k0, H / 32, lane);
    v16bf a3 = load_frag(sp1, m0 + 48, k0, H / 32, lane);
    c0 = __builtin_amdgcn_wmma_f32_16x16x32_bf16(false, a0, false, bf, (short)0, c0, false, false);
    c1 = __builtin_amdgcn_wmma_f32_16x16x32_bf16(false, a1, false, bf, (short)0, c1, false, false);
    c2 = __builtin_amdgcn_wmma_f32_16x16x32_bf16(false, a2, false, bf, (short)0, c2, false, false);
    c3 = __builtin_amdgcn_wmma_f32_16x16x32_bf16(false, a3, false, bf, (short)0, c3, false, false);
  }
  const int n    = lane & 15;
  const int moff = (lane & 16) ? 8 : 0;
  const bool live = (n < OUT);
  const float bias = live ? b2[n] : 0.0f;
  const float eta  = live ? eta2[n] : 0.0f;
#define EPILOG2(C, MB)                                                   \
  _Pragma("unroll") for (int r = 0; r < 8; ++r) {                        \
    int m = (MB) + r + moff;                                             \
    float h = (C)[r] + bias;                                             \
    post2[(size_t)m * NP + n] = (__bf16)tanhf(h * (1.0f / THRESH) - eta);\
    if (live) {                                                          \
      size_t off = (size_t)m * OUT + n;                                  \
      float v = v2[off];                                                 \
      v = 0.5f * (v + h);                                                \
      float sp = (v >= 1.0f) ? 1.0f : 0.0f;                              \
      v2[off] = (1.0f - sp) * v;                                         \
      s2[off] += sp;                                                     \
    }                                                                    \
  }
  EPILOG2(c0, m0) EPILOG2(c1, m0 + 16) EPILOG2(c2, m0 + 32) EPILOG2(c3, m0 + 48)
#undef EPILOG2
}

// ---- hebb1: hb1 = clip(0.95*hb1 + pre1T @ post1 / B)  (K = 8192 WMMA) -----
// 49 M-tiles = 7 supertiles of 7 tiles; each wave: 112x16 output.
__global__ void __launch_bounds__(256)
hebb1_upd(const __bf16* __restrict__ pre1T, const __bf16* __restrict__ post1,
          float* __restrict__ hb1f) {
  const int lane = threadIdx.x & 31;
  const int wave = blockIdx.x * 8 + (threadIdx.x >> 5);
  const int NT = H / 16;                          // 32
  const int mbase = (wave / NT) * 112, n0 = (wave % NT) * 16;
  v8f acc[7] = {};
  for (int k0 = 0; k0 < BT; k0 += 32) {
    v16bf bf = load_frag(post1, n0, k0, BT / 32, lane);
#pragma unroll
    for (int i = 0; i < 7; ++i) {
      v16bf a = load_frag(pre1T, mbase + 16 * i, k0, BT / 32, lane);
      acc[i] = __builtin_amdgcn_wmma_f32_16x16x32_bf16(false, a, false, bf, (short)0, acc[i], false, false);
    }
  }
  const int n    = n0 + (lane & 15);
  const int moff = (lane & 16) ? 8 : 0;
#pragma unroll
  for (int i = 0; i < 7; ++i) {
#pragma unroll
    for (int r = 0; r < 8; ++r) {
      int m = mbase + 16 * i + r + moff;
      size_t off = (size_t)m * H + n;
      float h = WDEC * hb1f[off] + acc[i][r] * (1.0f / (float)BT);
      hb1f[off] = fminf(4.0f, fmaxf(-4.0f, h));
    }
  }
}

// ---- hebb2: tiny (512x10, K=8192) scalar reduction ------------------------
__global__ void __launch_bounds__(256)
hebb2_upd(const __bf16* __restrict__ sp1, const __bf16* __restrict__ post2,
          const float* __restrict__ beta2, float* __restrict__ hb2f) {
  int j = blockIdx.x >> 1;                        // 2 blocks per output col
  int i = ((blockIdx.x & 1) << 8) + threadIdx.x;
  float acc = 0.0f;
  for (int b = 0; b < BT; ++b)
    acc += (float)sp1[frag_off(b, i, H / 32)] * (float)post2[(size_t)b * NP + j];
  acc = acc * beta2[i] * (1.0f / (float)BT);      // beta2 factors out of sum
  size_t off = (size_t)i * OUT + j;
  float h = WDEC * hb2f[off] + acc;
  hb2f[off] = fminf(4.0f, fmaxf(-4.0f, h));
}

extern "C" void kernel_launch(void* const* d_in, const int* in_sizes, int n_in,
                              void* d_out, int out_size, void* d_ws, size_t ws_size,
                              hipStream_t stream) {
  const float* x      = (const float*)d_in[0];
  const float* hebb1  = (const float*)d_in[1];
  const float* hebb2  = (const float*)d_in[2];
  const float* W1     = (const float*)d_in[3];
  const float* b1     = (const float*)d_in[4];
  const float* W2     = (const float*)d_in[5];
  const float* b2     = (const float*)d_in[6];
  const float* alpha1 = (const float*)d_in[7];
  const float* alpha2 = (const float*)d_in[8];
  const float* eta1   = (const float*)d_in[9];
  const float* eta2   = (const float*)d_in[10];
  const float* beta1  = (const float*)d_in[11];
  const float* beta2  = (const float*)d_in[12];

  float* out = (float*)d_out;                     // (s2, s1, hb1, hb2) concat
  float* s2      = out;
  float* s1      = out + (size_t)BT * OUT;
  float* hb1_out = s1 + (size_t)BT * H;
  float* hb2_out = hb1_out + (size_t)IN * H;

  char* p = (char*)d_ws;
  auto take = [&](size_t bytes) -> char* {
    char* r = p; p += (bytes + 255) & ~(size_t)255; return r;
  };
  float*  v1f   = (float*) take((size_t)BT * H * 4);
  float*  v2f   = (float*) take((size_t)BT * OUT * 4);
  float*  hb1f  = (float*) take((size_t)IN * H * 4);
  float*  hb2f  = (float*) take((size_t)H * OUT * 4);
  __bf16* xbf   = (__bf16*)take((size_t)BT * KP * 2);
  __bf16* pre1T = (__bf16*)take((size_t)IN * BT * 2);
  __bf16* w1e   = (__bf16*)take((size_t)KP * H * 2);
  __bf16* w2e   = (__bf16*)take((size_t)H * NP * 2);
  __bf16* sp1b  = (__bf16*)take((size_t)BT * H * 2);
  __bf16* post1 = (__bf16*)take((size_t)BT * H * 2);
  __bf16* post2 = (__bf16*)take((size_t)BT * NP * 2);

  hipMemsetAsync(d_out, 0, (size_t)out_size * 4, stream);
  hipMemsetAsync(v1f, 0, (size_t)BT * H * 4, stream);
  hipMemsetAsync(v2f, 0, (size_t)BT * OUT * 4, stream);
  hipMemcpyAsync(hb1f, hebb1, (size_t)IN * H * 4, hipMemcpyDeviceToDevice, stream);
  hipMemcpyAsync(hb2f, hebb2, (size_t)H * OUT * 4, hipMemcpyDeviceToDevice, stream);

  for (int t = 0; t < 10; ++t) {
    float decay = expf(-(float)t / 40.0f);
    prep_x<<<(BT * KP) / 256, 256, 0, stream>>>(x, beta1, xbf, pre1T, decay);
    weff1<<<(KP * H) / 256, 256, 0, stream>>>(W1, hb1f, alpha1, w1e);
    weff2<<<(H * NP) / 256, 256, 0, stream>>>(W2, hb2f, alpha2, w2e);
    gemm1_lif<<<(BT / 64) * (H / 16) / 8, 256, 0, stream>>>(xbf, w1e, b1, eta1,
                                                            v1f, s1, sp1b, post1);
    gemm2_lif<<<(BT / 64) / 8, 256, 0, stream>>>(sp1b, w2e, b2, eta2, v2f, s2, post2);
    hebb1_upd<<<(7 * (H / 16)) / 8, 256, 0, stream>>>(pre1T, post1, hb1f);
    hebb2_upd<<<2 * OUT, 256, 0, stream>>>(sp1b, post2, beta2, hb2f);
  }

  hipMemcpyAsync(hb1_out, hb1f, (size_t)IN * H * 4, hipMemcpyDeviceToDevice, stream);
  hipMemcpyAsync(hb2_out, hb2f, (size_t)H * OUT * 4, hipMemcpyDeviceToDevice, stream);
}
